// HMAEAttention_8461085573151
// MI455X (gfx1250) — compile-verified, matrix operand order checked by reading, not run on hardware
//
#include <hip/hip_runtime.h>

// ---------------------------------------------------------------------------
// HMAE attention block for MI455X (gfx1250, wave32, WMMA 16x16x32 f16)
// x:[2,2048,1024] -> qkv -> rope -> flash attention (async-LDS K/V) -> proj
// ---------------------------------------------------------------------------

typedef __attribute__((ext_vector_type(16))) _Float16 v16h;
typedef __attribute__((ext_vector_type(8)))  _Float16 hvec8;
typedef __attribute__((ext_vector_type(8)))  float    v8f;
typedef __attribute__((ext_vector_type(4)))  int      v4i;

#define BATCH 2
#define SEQ   2048
#define CDIM  1024
#define HEADS 16
#define HDIM  64
#define TOKENS (BATCH * SEQ)       // 4096
#define NQKV  (3 * CDIM)           // 3072
#define ATT_SCALE 0.125f           // 64^-0.5

// ---- async global->LDS copy (gfx1250), with safe fallback ------------------
#if defined(__gfx1250__) && __has_builtin(__builtin_amdgcn_global_load_async_to_lds_b128)
#define HAS_ASYNC_LDS 1
#endif

__device__ __forceinline__ void async_cp16(void* lds, const void* g) {
#ifdef HAS_ASYNC_LDS
  __builtin_amdgcn_global_load_async_to_lds_b128(
      (__attribute__((address_space(1))) v4i*)(uintptr_t)g,
      (__attribute__((address_space(3))) v4i*)(uintptr_t)lds,
      0, 0);
#else
  *(hvec8*)lds = *(const hvec8*)g;     // synchronous fallback
#endif
}

__device__ __forceinline__ void wait_async_lds() {
#ifdef HAS_ASYNC_LDS
#if __has_builtin(__builtin_amdgcn_s_wait_asynccnt)
  __builtin_amdgcn_s_wait_asynccnt(0);
#else
  asm volatile("s_wait_asynccnt 0" ::: "memory");
#endif
#endif
}

// A/B fragment load from row-major f16 matrix (ld = row stride in elements).
// Per CDNA5 ISA: lane (0-15) holds row (lane&15), K = {0..7,16..23};
// lane (16-31) holds same rows, K = {8..15,24..31}.
__device__ __forceinline__ v16h load_frag(const _Float16* base, int ld,
                                          int row0, int k0, int lane) {
  int r  = lane & 15;
  int kb = (lane >> 4) << 3;  // 0 or 8
  const _Float16* p = base + (size_t)(row0 + r) * ld + (k0 + kb);
  hvec8 lo = *(const hvec8*)(p);
  hvec8 hi = *(const hvec8*)(p + 16);
  return __builtin_shufflevector(lo, hi, 0, 1, 2, 3, 4, 5, 6, 7,
                                 8, 9, 10, 11, 12, 13, 14, 15);
}

__device__ __forceinline__ v8f wmma_f16(v16h a, v16h b, v8f c) {
  return __builtin_amdgcn_wmma_f32_16x16x32_f16(false, a, false, b,
                                                (short)0, c, false, false);
}

// ---------------------------------------------------------------------------
// fp32 -> f16 convert (grid-stride)
// ---------------------------------------------------------------------------
__global__ void cvt_f32_f16_kernel(const float* __restrict__ src,
                                   _Float16* __restrict__ dst, int n) {
  for (int i = blockIdx.x * blockDim.x + threadIdx.x; i < n;
       i += gridDim.x * blockDim.x)
    dst[i] = (_Float16)src[i];
}

// ---------------------------------------------------------------------------
// QKV GEMM: Y[m,n] = x[m,:] . qkv_w[n,:] + b[n],  M=4096, N=3072, K=1024
// block = 256 thr (8 waves), tile 64(M) x 128(N); wave = 32x32 (2x2 frags)
// epilogue scatters q,k -> [B,H,N,D], v -> [B,H,D,N] (transposed) as f16
// ---------------------------------------------------------------------------
__global__ void __launch_bounds__(256)
qkv_gemm_kernel(const _Float16* __restrict__ xh,
                const _Float16* __restrict__ wh,
                const float* __restrict__ bias,
                _Float16* __restrict__ qh,
                _Float16* __restrict__ kh,
                _Float16* __restrict__ vT) {
  const int lane = threadIdx.x & 31;
  const int wid  = threadIdx.x >> 5;
  const int wm   = wid & 1;        // 0..1
  const int wn   = wid >> 1;       // 0..3
  const int row_base = blockIdx.y * 64 + wm * 32;
  const int col_base = blockIdx.x * 128 + wn * 32;

  v8f acc00 = {}, acc01 = {}, acc10 = {}, acc11 = {};

#pragma unroll 4
  for (int k0 = 0; k0 < CDIM; k0 += 32) {
    // stream-ahead prefetch (global_prefetch_b8), 256 elements = 8 k-steps ahead
    if (k0 + 256 < CDIM) {
      __builtin_prefetch(xh + (size_t)(row_base + (lane & 15)) * CDIM + k0 + 256, 0, 1);
      __builtin_prefetch(wh + (size_t)(col_base + (lane & 15)) * CDIM + k0 + 256, 0, 1);
    }
    v16h a0 = load_frag(xh, CDIM, row_base,      k0, lane);
    v16h a1 = load_frag(xh, CDIM, row_base + 16, k0, lane);
    v16h b0 = load_frag(wh, CDIM, col_base,      k0, lane);
    v16h b1 = load_frag(wh, CDIM, col_base + 16, k0, lane);
    acc00 = wmma_f16(a0, b0, acc00);
    acc01 = wmma_f16(a0, b1, acc01);
    acc10 = wmma_f16(a1, b0, acc10);
    acc11 = wmma_f16(a1, b1, acc11);
  }

  const int nlo = lane & 15;
  const int mo  = (lane >> 4) << 3;   // 0 or 8
  v8f accs[2][2] = {{acc00, acc01}, {acc10, acc11}};

#pragma unroll
  for (int mi = 0; mi < 2; ++mi)
#pragma unroll
    for (int ni = 0; ni < 2; ++ni)
#pragma unroll
      for (int i = 0; i < 8; ++i) {
        int m = row_base + mi * 16 + mo + i;   // token index (0..4095)
        int n = col_base + ni * 16 + nlo;      // qkv feature (0..3071)
        float v = accs[mi][ni][i] + bias[n];
        int b = m >> 11;
        int tok = m & (SEQ - 1);
        if (n < CDIM) {
          int h = n >> 6, d = n & 63;
          qh[(((size_t)(b * HEADS + h)) * SEQ + tok) * HDIM + d] = (_Float16)v;
        } else if (n < 2 * CDIM) {
          int n2 = n - CDIM;
          int h = n2 >> 6, d = n2 & 63;
          kh[(((size_t)(b * HEADS + h)) * SEQ + tok) * HDIM + d] = (_Float16)v;
        } else {
          int n2 = n - 2 * CDIM;
          int h = n2 >> 6, d = n2 & 63;
          vT[(((size_t)(b * HEADS + h)) * HDIM + d) * SEQ + tok] = (_Float16)v;
        }
      }
}

// ---------------------------------------------------------------------------
// RoPE on q and k (f16 in/out, f32 math). Each thread owns a (d, d+32) pair
// of one (b,h,tok) vector -> race-free in-place update. threads = 2M
// ---------------------------------------------------------------------------
__global__ void rope_kernel(_Float16* __restrict__ qh,
                            _Float16* __restrict__ kh,
                            const float* __restrict__ rcos,
                            const float* __restrict__ rsin) {
  int idx = blockIdx.x * blockDim.x + threadIdx.x;  // 21 bits
  int d   = idx & 31;
  int tok = (idx >> 5) & (SEQ - 1);
  int bh  = idx >> 16;                              // 0..31
  size_t base = ((size_t)bh * SEQ + tok) * HDIM;
  float c1 = rcos[tok * HDIM + d];
  float s1 = rsin[tok * HDIM + d];
  float c2 = rcos[tok * HDIM + d + 32];
  float s2 = rsin[tok * HDIM + d + 32];

  float q1 = (float)qh[base + d], q2 = (float)qh[base + d + 32];
  qh[base + d]      = (_Float16)(q1 * c1 - q2 * s1);   // rot = -x2
  qh[base + d + 32] = (_Float16)(q2 * c2 + q1 * s2);   // rot = +x1

  float k1 = (float)kh[base + d], k2 = (float)kh[base + d + 32];
  kh[base + d]      = (_Float16)(k1 * c1 - k2 * s1);
  kh[base + d + 32] = (_Float16)(k2 * c2 + k1 * s2);
}

// ---------------------------------------------------------------------------
// Flash attention: grid = (SEQ/64, B*H), block = 128 (4 waves).
// Each wave: 16 query rows, online softmax over 2048 keys in steps of 32.
// K tile (32x64) and V^T tile (64x32) are block-shared: staged into LDS with
// double-buffered async copies (GLOBAL_LOAD_ASYNC_TO_LDS_B128 + ASYNCcnt),
// removing the 4x cross-wave duplicate global loads.
// ---------------------------------------------------------------------------
#define KT_LD 72   // 64 + 8 halfs pad
#define VT_LD 40   // 32 + 8 halfs pad
#define NIT  (SEQ / 32)

__global__ void __launch_bounds__(128)
attn_kernel(const _Float16* __restrict__ qh,
            const _Float16* __restrict__ kh,
            const _Float16* __restrict__ vT,
            _Float16* __restrict__ attnout) {
  const int tid  = threadIdx.x;
  const int lane = tid & 31;
  const int wid  = tid >> 5;
  const int bh   = blockIdx.y;             // b*H + h
  const int qrow = blockIdx.x * 64 + wid * 16;
  const int nlo  = lane & 15;
  const int mo   = (lane >> 4) << 3;

  const _Float16* qb = qh + (size_t)bh * SEQ * HDIM;
  const _Float16* kb = kh + (size_t)bh * SEQ * HDIM;
  const _Float16* vb = vT + (size_t)bh * HDIM * SEQ;

  __shared__ _Float16 kt[2][32 * KT_LD];   // [key][d]
  __shared__ _Float16 vt[2][64 * VT_LD];   // [d][key]
  __shared__ _Float16 pbuf[4][16 * 40];    // per-wave P relayout
  _Float16* pw = &pbuf[wid][0];

  // stage one 32-key step: 256 16B chunks for K, 256 for V, 128 threads
  auto stage = [&](int buf, int j0) {
#pragma unroll
    for (int s = 0; s < 2; ++s) {
      int c  = tid + s * 128;
      int rk = c >> 3, ck = (c & 7) * 8;            // K: 8 chunks per 64-h row
      async_cp16(&kt[buf][rk * KT_LD + ck],
                 kb + (size_t)(j0 + rk) * HDIM + ck);
      int rv = c >> 2, cv = (c & 3) * 8;            // V: 4 chunks per 32-h row
      async_cp16(&vt[buf][rv * VT_LD + cv],
                 vb + (size_t)rv * SEQ + j0 + cv);
    }
  };

  v16h qa0 = load_frag(qb, HDIM, qrow, 0,  lane);
  v16h qa1 = load_frag(qb, HDIM, qrow, 32, lane);

  float mrow[8], lrow[8];
  v8f o0 = {}, o1 = {}, o2 = {}, o3 = {};
#pragma unroll
  for (int i = 0; i < 8; ++i) { mrow[i] = -1e30f; lrow[i] = 0.0f; }

  stage(0, 0);

  for (int it = 0; it < NIT; ++it) {
    const int j0 = it * 32;
    wait_async_lds();      // our staged chunks for buf[it&1] have landed
    __syncthreads();       // every wave's chunks have landed
    if (it + 1 < NIT) stage((it + 1) & 1, j0 + 32);

    const _Float16* ktc = &kt[it & 1][0];
    const _Float16* vtc = &vt[it & 1][0];

    v16h kb0a = load_frag(ktc, KT_LD, 0,  0,  lane);
    v16h kb0b = load_frag(ktc, KT_LD, 0,  32, lane);
    v16h kb1a = load_frag(ktc, KT_LD, 16, 0,  lane);
    v16h kb1b = load_frag(ktc, KT_LD, 16, 32, lane);

    v8f s0 = {}, s1 = {};
    s0 = wmma_f16(qa0, kb0a, s0);
    s0 = wmma_f16(qa1, kb0b, s0);
    s1 = wmma_f16(qa0, kb1a, s1);
    s1 = wmma_f16(qa1, kb1b, s1);
    s0 *= ATT_SCALE;
    s1 *= ATT_SCALE;

    float p0[8], p1[8], alpha[8];
#pragma unroll
    for (int i = 0; i < 8; ++i) {
      float mx = fmaxf(s0[i], s1[i]);
#pragma unroll
      for (int off = 8; off; off >>= 1)   // xor<=8 stays in 16-lane row group
        mx = fmaxf(mx, __shfl_xor(mx, off, 32));
      float mnew = fmaxf(mrow[i], mx);
      alpha[i] = __expf(mrow[i] - mnew);
      p0[i] = __expf(s0[i] - mnew);
      p1[i] = __expf(s1[i] - mnew);
      float rs = p0[i] + p1[i];
#pragma unroll
      for (int off = 8; off; off >>= 1)
        rs += __shfl_xor(rs, off, 32);
      lrow[i] = lrow[i] * alpha[i] + rs;
      mrow[i] = mnew;
    }
#pragma unroll
    for (int i = 0; i < 8; ++i) {
      o0[i] *= alpha[i]; o1[i] *= alpha[i];
      o2[i] *= alpha[i]; o3[i] *= alpha[i];
    }

    // P (C-layout) -> LDS -> A-layout fragment (per-wave buffer, LDS in-order)
#pragma unroll
    for (int i = 0; i < 8; ++i) {
      pw[(mo + i) * 40 + nlo]      = (_Float16)p0[i];
      pw[(mo + i) * 40 + 16 + nlo] = (_Float16)p1[i];
    }
    v16h pa = load_frag(pw, 40, 0, 0, lane);

    v16h vf0 = load_frag(vtc, VT_LD, 0,  0, lane);
    v16h vf1 = load_frag(vtc, VT_LD, 16, 0, lane);
    v16h vf2 = load_frag(vtc, VT_LD, 32, 0, lane);
    v16h vf3 = load_frag(vtc, VT_LD, 48, 0, lane);
    o0 = wmma_f16(pa, vf0, o0);
    o1 = wmma_f16(pa, vf1, o1);
    o2 = wmma_f16(pa, vf2, o2);
    o3 = wmma_f16(pa, vf3, o3);
  }

  // normalize + store to [token, C] f16 (C = H*D, head-major columns)
  const int b = bh >> 4, h = bh & 15;
  v8f oo[4] = {o0, o1, o2, o3};
#pragma unroll
  for (int g = 0; g < 4; ++g)
#pragma unroll
    for (int i = 0; i < 8; ++i) {
      float val = oo[g][i] / lrow[i];
      size_t row = (size_t)b * SEQ + (qrow + mo + i);
      int col = h * HDIM + g * 16 + nlo;
      attnout[row * CDIM + col] = (_Float16)val;
    }
}

// ---------------------------------------------------------------------------
// Proj GEMM: out[m,n] = attn[m,:] . proj_w[n,:] + b[n], fp32 out
// ---------------------------------------------------------------------------
__global__ void __launch_bounds__(256)
proj_gemm_kernel(const _Float16* __restrict__ ah,
                 const _Float16* __restrict__ wh,
                 const float* __restrict__ bias,
                 float* __restrict__ out) {
  const int lane = threadIdx.x & 31;
  const int wid  = threadIdx.x >> 5;
  const int wm   = wid & 1;
  const int wn   = wid >> 1;
  const int row_base = blockIdx.y * 64 + wm * 32;
  const int col_base = blockIdx.x * 128 + wn * 32;

  v8f acc00 = {}, acc01 = {}, acc10 = {}, acc11 = {};
#pragma unroll 4
  for (int k0 = 0; k0 < CDIM; k0 += 32) {
    if (k0 + 256 < CDIM) {
      __builtin_prefetch(ah + (size_t)(row_base + (lane & 15)) * CDIM + k0 + 256, 0, 1);
      __builtin_prefetch(wh + (size_t)(col_base + (lane & 15)) * CDIM + k0 + 256, 0, 1);
    }
    v16h a0 = load_frag(ah, CDIM, row_base,      k0, lane);
    v16h a1 = load_frag(ah, CDIM, row_base + 16, k0, lane);
    v16h b0 = load_frag(wh, CDIM, col_base,      k0, lane);
    v16h b1 = load_frag(wh, CDIM, col_base + 16, k0, lane);
    acc00 = wmma_f16(a0, b0, acc00);
    acc01 = wmma_f16(a0, b1, acc01);
    acc10 = wmma_f16(a1, b0, acc10);
    acc11 = wmma_f16(a1, b1, acc11);
  }

  const int nlo = lane & 15;
  const int mo  = (lane >> 4) << 3;
  v8f accs[2][2] = {{acc00, acc01}, {acc10, acc11}};
#pragma unroll
  for (int mi = 0; mi < 2; ++mi)
#pragma unroll
    for (int ni = 0; ni < 2; ++ni)
#pragma unroll
      for (int i = 0; i < 8; ++i) {
        int m = row_base + mi * 16 + mo + i;
        int n = col_base + ni * 16 + nlo;
        out[(size_t)m * CDIM + n] = accs[mi][ni][i] + bias[n];
      }
}

// ---------------------------------------------------------------------------
extern "C" void kernel_launch(void* const* d_in, const int* in_sizes, int n_in,
                              void* d_out, int out_size, void* d_ws, size_t ws_size,
                              hipStream_t stream) {
  const float* x        = (const float*)d_in[0];
  const float* rope_cos = (const float*)d_in[1];
  const float* rope_sin = (const float*)d_in[2];
  const float* qkv_w    = (const float*)d_in[3];
  const float* qkv_b    = (const float*)d_in[4];
  const float* proj_w   = (const float*)d_in[5];
  const float* proj_b   = (const float*)d_in[6];
  float* out = (float*)d_out;

  // workspace carve (f16 buffers)
  char* ws = (char*)d_ws;
  const size_t n_x    = (size_t)TOKENS * CDIM;   // 4.19M
  const size_t n_wq   = (size_t)NQKV * CDIM;     // 3.15M
  const size_t n_wp   = (size_t)CDIM * CDIM;     // 1.05M
  const size_t n_head = (size_t)BATCH * HEADS * SEQ * HDIM;  // 4.19M

  _Float16* xh    = (_Float16*)ws;                 ws += n_x * 2;
  _Float16* wqkvh = (_Float16*)ws;                 ws += n_wq * 2;
  _Float16* wprjh = (_Float16*)ws;                 ws += n_wp * 2;
  _Float16* qh    = (_Float16*)ws;                 ws += n_head * 2;
  _Float16* kh    = (_Float16*)ws;                 ws += n_head * 2;
  _Float16* vT    = (_Float16*)ws;                 ws += n_head * 2;
  _Float16* attnh = (_Float16*)ws;                 ws += n_x * 2;
  (void)ws_size; (void)n_in; (void)in_sizes; (void)out_size;

  // 1) convert to f16
  cvt_f32_f16_kernel<<<2048, 256, 0, stream>>>(x, xh, (int)n_x);
  cvt_f32_f16_kernel<<<2048, 256, 0, stream>>>(qkv_w, wqkvh, (int)n_wq);
  cvt_f32_f16_kernel<<<1024, 256, 0, stream>>>(proj_w, wprjh, (int)n_wp);

  // 2) QKV GEMM + scatter (v transposed)
  qkv_gemm_kernel<<<dim3(NQKV / 128, TOKENS / 64), 256, 0, stream>>>(
      xh, wqkvh, qkv_b, qh, kh, vT);

  // 3) RoPE on q, k
  rope_kernel<<<(BATCH * HEADS * SEQ * 32) / 256, 256, 0, stream>>>(
      qh, kh, rope_cos, rope_sin);

  // 4) flash attention (async-LDS staged K/V)
  attn_kernel<<<dim3(SEQ / 64, BATCH * HEADS), 128, 0, stream>>>(
      qh, kh, vT, attnh);

  // 5) output projection (fp32 out)
  proj_gemm_kernel<<<dim3(CDIM / 128, TOKENS / 64), 256, 0, stream>>>(
      attnh, wprjh, proj_b, out);
}
